// PointNetFeaturePropagation_2946347565086
// MI455X (gfx1250) — matrix operand order repaired and checked by reading, not compiled, not run on hardware
//
#include <hip/hip_runtime.h>
#include <hip/hip_bf16.h>

// ---- problem constants ----
#define B_   8
#define N_   8192
#define S_   2048
#define D1_  128
#define D2_  256
#define K1_  (D1_ + D2_)   // 384
#define C1_  256           // hidden channels after conv1
#define C2_  128           // channels after conv2
#define RT_  (B_ * N_)     // 65536 total rows
#define BN_EPS 1e-5f

typedef __attribute__((ext_vector_type(16))) _Float16 v16h;
typedef __attribute__((ext_vector_type(8)))  _Float16 v8h;
typedef __attribute__((ext_vector_type(8)))  float    v8f;

// ---------------------------------------------------------------------------
// Kernel 1: 3-nearest-neighbor search. xyz2 tile (24KB) staged in LDS.
// ---------------------------------------------------------------------------
__global__ __launch_bounds__(256) void knn3_kernel(
    const float* __restrict__ xyz1, const float* __restrict__ xyz2,
    int* __restrict__ nidx, float* __restrict__ nwgt)
{
    __shared__ float sx[S_], sy[S_], sz[S_];
    const int b = blockIdx.y;
    const int n = blockIdx.x * 256 + threadIdx.x;

    const float* x2 = xyz2 + (size_t)b * 3 * S_;
    for (int s = threadIdx.x; s < S_; s += 256) {
        sx[s] = x2[s];
        sy[s] = x2[S_ + s];
        sz[s] = x2[2 * S_ + s];
    }
    __syncthreads();

    const float* x1 = xyz1 + (size_t)b * 3 * N_;
    const float px = x1[n], py = x1[N_ + n], pz = x1[2 * N_ + n];

    float d0 = 3.4e38f, d1 = 3.4e38f, d2 = 3.4e38f;
    int   i0 = 0, i1 = 0, i2 = 0;
    for (int s = 0; s < S_; ++s) {
        const float dx = px - sx[s], dy = py - sy[s], dz = pz - sz[s];
        const float d = dx * dx + dy * dy + dz * dz;
        if (d < d0)      { d2 = d1; i2 = i1; d1 = d0; i1 = i0; d0 = d; i0 = s; }
        else if (d < d1) { d2 = d1; i2 = i1; d1 = d;  i1 = s; }
        else if (d < d2) { d2 = d;  i2 = s; }
    }
    d0 = fmaxf(d0, 1e-10f); d1 = fmaxf(d1, 1e-10f); d2 = fmaxf(d2, 1e-10f);
    const float w0 = 1.f / d0, w1 = 1.f / d1, w2 = 1.f / d2;
    const float inv = 1.f / (w0 + w1 + w2);

    const size_t r = (size_t)b * N_ + n;
    nidx[r * 3 + 0] = i0; nidx[r * 3 + 1] = i1; nidx[r * 3 + 2] = i2;
    nwgt[r * 3 + 0] = w0 * inv; nwgt[r * 3 + 1] = w1 * inv; nwgt[r * 3 + 2] = w2 * inv;
}

// ---------------------------------------------------------------------------
// fp32 -> f16 conversion (weights)
// ---------------------------------------------------------------------------
__global__ void f32_to_f16_kernel(const float* __restrict__ in,
                                  _Float16* __restrict__ out, int n)
{
    const int i = blockIdx.x * blockDim.x + threadIdx.x;
    if (i < n) out[i] = (_Float16)in[i];
}

// ---------------------------------------------------------------------------
// Kernel 2: build X = [points1^T | interp(points2)] as f16, row-major [RT, 384]
// ---------------------------------------------------------------------------
__global__ void build_x_kernel(const float* __restrict__ p1,
                               const float* __restrict__ p2,
                               const int* __restrict__ nidx,
                               const float* __restrict__ nwgt,
                               _Float16* __restrict__ X)
{
    const size_t i = (size_t)blockIdx.x * blockDim.x + threadIdx.x; // r*K1 + c
    const int    c = (int)(i % K1_);
    const size_t r = i / K1_;
    const int    n = (int)(r % N_);
    const int    b = (int)(r / N_);

    float v;
    if (c < D1_) {
        v = p1[((size_t)b * D1_ + c) * N_ + n];
    } else {
        const int c2 = c - D1_;
        const float* f  = p2 + ((size_t)b * D2_ + c2) * S_;
        const int*   ip = nidx + r * 3;
        const float* wp = nwgt + r * 3;
        v = f[ip[0]] * wp[0] + f[ip[1]] * wp[1] + f[ip[2]] * wp[2];
    }
    X[i] = (_Float16)v;
}

// ---------------------------------------------------------------------------
// WMMA GEMM: Y[M,Nout] = A[M,K](f16) * W[Nout,K]^T(f16) + bias, f32 accum.
// Block = 256 threads = 8 waves, block tile 128(M) x 128(N).
// Weight slab (128 rows x K halves) staged once per block in LDS with a
// +8-half row pad (stride 196 dwords == 4 mod 64 -> conflict-free B reads).
// Wave tile 16(M) x 128(N): 8 v8f accumulators; A loaded from global per the
// ISA 16-bit A-fragment layout, B fragments read from LDS.
// ---------------------------------------------------------------------------
template<int K>
__global__ __launch_bounds__(256) void gemm_wmma_kernel(
    const _Float16* __restrict__ A, const _Float16* __restrict__ W,
    const float* __restrict__ bias, float* __restrict__ Y, int Nout)
{
    constexpr int LDK = K + 8;                 // padded LDS row stride (halves)
    __shared__ _Float16 sW[128 * LDK];

    const int tid  = threadIdx.x;
    const int lane = tid & 31;
    const int wave = tid >> 5;
    const int hi   = lane >> 4;                // half-wave select
    const int lr   = lane & 15;

    const int m0 = blockIdx.x * 128 + wave * 16;
    const int n0 = blockIdx.y * 128;

    // ---- cooperative load of weight slab into LDS (16B vectors) ----
    for (int row = wave; row < 128; row += 8) {
        const _Float16* src = W + (size_t)(n0 + row) * K;
        _Float16*       dst = sW + row * LDK;
        for (int k = lane * 8; k < K; k += 32 * 8)
            *(v8h*)(dst + k) = *(const v8h*)(src + k);
    }
    __syncthreads();

    v8f acc[8];
    const v8f vzero = {};
#pragma unroll
    for (int t = 0; t < 8; ++t) acc[t] = vzero;

    const _Float16* arow = A + (size_t)(m0 + lr) * K + hi * 8;
    const _Float16* swb  = sW + lr * LDK + hi * 8;

    for (int k0 = 0; k0 < K; k0 += 32) {
        __builtin_prefetch(arow + k0 + 64, 0, 1);     // global_prefetch_b8
        const v8h alo = *(const v8h*)(arow + k0);
        const v8h ahi = *(const v8h*)(arow + k0 + 16);
        v16h a;
#pragma unroll
        for (int i = 0; i < 8; ++i) { a[i] = alo[i]; a[8 + i] = ahi[i]; }

#pragma unroll
        for (int t = 0; t < 8; ++t) {
            const _Float16* bp = swb + t * 16 * LDK + k0;
            const v8h blo = *(const v8h*)(bp);        // ds_load_b128
            const v8h bhi = *(const v8h*)(bp + 16);   // ds_load_b128
            v16h b;
#pragma unroll
            for (int i = 0; i < 8; ++i) { b[i] = blo[i]; b[8 + i] = bhi[i]; }
            acc[t] = __builtin_amdgcn_wmma_f32_16x16x32_f16(
                false, a, false, b, (short)0, acc[t], false, false);
        }
    }

#pragma unroll
    for (int t = 0; t < 8; ++t) {
        const int   n  = n0 + t * 16 + lr;
        const float bs = bias[n];
#pragma unroll
        for (int r = 0; r < 8; ++r) {
            const int m = m0 + hi * 8 + r;   // C/D layout: VGPR r -> M=r (+8 hi half)
            Y[(size_t)m * Nout + n] = acc[t][r] + bs;
        }
    }
}

// ---------------------------------------------------------------------------
// Per-channel sum / sumsq reduction over rows (coalesced row-major reads).
// blockDim.x == C; each block handles rowsPerBlock rows.
// ---------------------------------------------------------------------------
__global__ void col_stats_kernel(const float* __restrict__ Y,
                                 float* __restrict__ sum, float* __restrict__ sumsq,
                                 int C, int rowsPerBlock)
{
    const int c = threadIdx.x;
    const size_t r0 = (size_t)blockIdx.x * rowsPerBlock;
    float s = 0.f, s2 = 0.f;
    for (int i = 0; i < rowsPerBlock; ++i) {
        const float v = Y[(r0 + i) * C + c];
        s += v; s2 += v * v;
    }
    atomicAdd(&sum[c], s);
    atomicAdd(&sumsq[c], s2);
}

// Fold BN stats into per-channel affine: y = x*a + c
__global__ void bn_params_kernel(const float* __restrict__ sum,
                                 const float* __restrict__ sumsq,
                                 const float* __restrict__ g,
                                 const float* __restrict__ be,
                                 float* __restrict__ aOut, float* __restrict__ cOut,
                                 float invM)
{
    const int c = threadIdx.x;
    const float mean = sum[c] * invM;
    const float var  = sumsq[c] * invM - mean * mean;
    const float sc   = g[c] * rsqrtf(var + BN_EPS);
    aOut[c] = sc;
    cOut[c] = be[c] - mean * sc;
}

// BN affine + ReLU, downconvert to f16 (feeds next GEMM)
__global__ void bn_relu_f16_kernel(const float* __restrict__ Y,
                                   const float* __restrict__ a,
                                   const float* __restrict__ cc,
                                   _Float16* __restrict__ X, int C)
{
    const size_t i = (size_t)blockIdx.x * blockDim.x + threadIdx.x;
    const int c = (int)(i % C);
    const float v = fmaxf(Y[i] * a[c] + cc[c], 0.f);
    X[i] = (_Float16)v;
}

// ---------------------------------------------------------------------------
// BN affine + ReLU + transpose to [B, C2, N] fp32 via LDS tile (32n x 128c).
// Both phases coalesced; +1 pad keeps LDS banks conflict-free.
// ---------------------------------------------------------------------------
__global__ __launch_bounds__(256) void bn_relu_out_kernel(
    const float* __restrict__ Y, const float* __restrict__ a,
    const float* __restrict__ cc, float* __restrict__ out)
{
    __shared__ float t[32][C2_ + 1];
    const int tid = threadIdx.x;
    const int b   = blockIdx.y;
    const int n0  = blockIdx.x * 32;

    // load: Y[(b*N + n0+n)*C2 + c], c fastest across threads (coalesced)
#pragma unroll
    for (int i = 0; i < 16; ++i) {
        const int e = tid + i * 256;            // e = n*128 + c
        const int n = e >> 7, c = e & (C2_ - 1);
        const float v = Y[((size_t)b * N_ + (n0 + n)) * C2_ + c] * a[c] + cc[c];
        t[n][c] = fmaxf(v, 0.f);
    }
    __syncthreads();

    // store: out[(b*C2 + c)*N + n0+n], n fastest across threads (coalesced)
#pragma unroll
    for (int i = 0; i < 16; ++i) {
        const int e = tid + i * 256;            // e = c*32 + n
        const int n = e & 31, c = e >> 5;
        out[((size_t)b * C2_ + c) * N_ + n0 + n] = t[n][c];
    }
}

// ---------------------------------------------------------------------------
extern "C" void kernel_launch(void* const* d_in, const int* in_sizes, int n_in,
                              void* d_out, int out_size, void* d_ws, size_t ws_size,
                              hipStream_t stream)
{
    const float* xyz1    = (const float*)d_in[0];
    const float* xyz2    = (const float*)d_in[1];
    const float* points1 = (const float*)d_in[2];
    const float* points2 = (const float*)d_in[3];
    const float* w1      = (const float*)d_in[4];
    const float* b1      = (const float*)d_in[5];
    const float* g1      = (const float*)d_in[6];
    const float* be1     = (const float*)d_in[7];
    const float* w2      = (const float*)d_in[8];
    const float* b2      = (const float*)d_in[9];
    const float* g2      = (const float*)d_in[10];
    const float* be2     = (const float*)d_in[11];
    float* out = (float*)d_out;

    // ---- workspace layout (256B aligned) ----
    char* ws = (char*)d_ws;
    size_t off = 0;
    auto alloc = [&](size_t bytes) -> void* {
        void* p = ws + off;
        off += (bytes + 255) & ~(size_t)255;
        return p;
    };
    int*      nidx = (int*)     alloc((size_t)RT_ * 3 * sizeof(int));
    float*    nwgt = (float*)   alloc((size_t)RT_ * 3 * sizeof(float));
    _Float16* Xh   = (_Float16*)alloc((size_t)RT_ * K1_ * sizeof(_Float16));
    _Float16* W1h  = (_Float16*)alloc((size_t)C1_ * K1_ * sizeof(_Float16));
    _Float16* W2h  = (_Float16*)alloc((size_t)C2_ * C1_ * sizeof(_Float16));
    float*    Y1   = (float*)   alloc((size_t)RT_ * C1_ * sizeof(float));
    _Float16* X2h  = (_Float16*)alloc((size_t)RT_ * C1_ * sizeof(_Float16));
    float*    Y2   = (float*)   alloc((size_t)RT_ * C2_ * sizeof(float));
    float*    st1  = (float*)   alloc(2 * C1_ * sizeof(float)); // sum | sumsq
    float*    bnA1 = (float*)   alloc(C1_ * sizeof(float));
    float*    bnC1 = (float*)   alloc(C1_ * sizeof(float));
    float*    st2  = (float*)   alloc(2 * C2_ * sizeof(float));
    float*    bnA2 = (float*)   alloc(C2_ * sizeof(float));
    float*    bnC2 = (float*)   alloc(C2_ * sizeof(float));
    (void)ws_size; (void)in_sizes; (void)n_in; (void)out_size;

    // ---- 1) 3-NN search ----
    knn3_kernel<<<dim3(N_ / 256, B_), 256, 0, stream>>>(xyz1, xyz2, nidx, nwgt);

    // ---- 2) weights -> f16 ----
    f32_to_f16_kernel<<<(C1_ * K1_ + 255) / 256, 256, 0, stream>>>(w1, W1h, C1_ * K1_);
    f32_to_f16_kernel<<<(C2_ * C1_ + 255) / 256, 256, 0, stream>>>(w2, W2h, C2_ * C1_);

    // ---- 3) build concat input X (f16, [RT, 384]) ----
    build_x_kernel<<<(unsigned)(((size_t)RT_ * K1_) / 256), 256, 0, stream>>>(
        points1, points2, nidx, nwgt, Xh);

    // ---- 4) GEMM1: Y1 = X * W1^T + b1 ----
    hipMemsetAsync(st1, 0, 2 * C1_ * sizeof(float), stream);
    gemm_wmma_kernel<K1_><<<dim3(RT_ / 128, C1_ / 128), 256, 0, stream>>>(
        Xh, W1h, b1, Y1, C1_);

    // ---- 5) BN1 stats + affine+ReLU -> f16 ----
    col_stats_kernel<<<RT_ / 256, C1_, 0, stream>>>(Y1, st1, st1 + C1_, C1_, 256);
    bn_params_kernel<<<1, C1_, 0, stream>>>(st1, st1 + C1_, g1, be1, bnA1, bnC1,
                                            1.0f / (float)RT_);
    bn_relu_f16_kernel<<<(unsigned)(((size_t)RT_ * C1_) / 256), 256, 0, stream>>>(
        Y1, bnA1, bnC1, X2h, C1_);

    // ---- 6) GEMM2: Y2 = X2 * W2^T + b2 ----
    hipMemsetAsync(st2, 0, 2 * C2_ * sizeof(float), stream);
    gemm_wmma_kernel<C1_><<<dim3(RT_ / 128, C2_ / 128), 256, 0, stream>>>(
        X2h, W2h, b2, Y2, C2_);

    // ---- 7) BN2 stats + affine+ReLU -> transposed fp32 output ----
    col_stats_kernel<<<RT_ / 256, C2_, 0, stream>>>(Y2, st2, st2 + C2_, C2_, 256);
    bn_params_kernel<<<1, C2_, 0, stream>>>(st2, st2 + C2_, g2, be2, bnA2, bnC2,
                                            1.0f / (float)RT_);
    bn_relu_out_kernel<<<dim3(N_ / 32, B_), 256, 0, stream>>>(Y2, bnA2, bnC2, out);
}
